// EncoderDecoder_14267881357368
// MI455X (gfx1250) — matrix operand order repaired and checked by reading, not compiled, not run on hardware
//
#include <hip/hip_runtime.h>
#include <hip/hip_bf16.h>

typedef _Float16 half_t;
typedef _Float16 v16h __attribute__((ext_vector_type(16)));
typedef _Float16 v8h  __attribute__((ext_vector_type(8)));
typedef float    v8f  __attribute__((ext_vector_type(8)));
typedef int      v4i_t __attribute__((vector_size(16)));   // matches builtin param

#define CONV_FWD   0
#define CONV_TRANS 1
#define EPI_RELU_F16  0
#define EPI_LIN_F32   1
#define EPI_DRELU_F32 2

#ifndef __has_builtin
#define __has_builtin(x) 0
#endif

// CDNA5 async global->LDS staging (ASYNCcnt-tracked), guarded so that a
// toolchain without the builtin falls back to the VGPR-bounce path.
// Probe-confirmed signature: (AS1 int4* src, AS3 int4* dst, imm off, imm cpol).
#if defined(__gfx1250__) && __has_builtin(__builtin_amdgcn_global_load_async_to_lds_b128)
#define HAVE_ASYNC_LDS 1
#define AS1P(p) ((__attribute__((address_space(1))) v4i_t*)(p))
#define AS3P(p) ((__attribute__((address_space(3))) v4i_t*)(p))
#if __has_builtin(__builtin_amdgcn_s_wait_asynccnt)
#define ASYNC_WAIT() __builtin_amdgcn_s_wait_asynccnt(0)
#else
#define ASYNC_WAIT() asm volatile("s_wait_asynccnt 0" ::: "memory")
#endif
#else
#define HAVE_ASYNC_LDS 0
#endif

// ---------------------------------------------------------------------------
// Implicit-GEMM conv via V_WMMA_F32_16X16X32_F16, NHWC activations.
// Block = 128 threads = 4 wave32s. Block tile: 64 (Cout) x 64 (spatial cols).
// K is tap-major: step s -> (tap = s/CCH, chunk = s%CCH); within a 32-wide K
// chunk the K index IS the input channel (contiguous in NHWC), so B staging
// is two b128 transfers per thread. CIN is a template constant so CCH/Kpad/
// nsteps and the ragged-channel predicate fold at compile time.
// LDS double-buffered; staging of step s+1 (async global->LDS where
// available) overlaps WMMA on step s.
// ---------------------------------------------------------------------------
template <int MODE, int EPI, int CIN>
__global__ __launch_bounds__(128)
void conv_wmma_k(const half_t* __restrict__ x,     // NHWC f16 [N,Hin,Win,CIN]
                 const half_t* __restrict__ wpk,   // [Cout][Kpad] f16 tap-major
                 const float*  __restrict__ bias,  // [Cout]
                 half_t* __restrict__ out16,       // NHWC f16
                 float*  __restrict__ out32,       // NHWC f32 (LIN) / NCHW f32 (DRELU)
                 int Hin, int Win,
                 int Cout, int Hout, int Wout)
{
  constexpr int CCH    = (CIN + 31) / 32;      // 1, 2 or 4 -> shifts
  constexpr int KP     = 9 * CCH * 32;
  constexpr int NSTEPS = 9 * CCH;
  constexpr bool FULLC = (CIN % 32) == 0;      // channel-aligned fast path

  __shared__ half_t bT[2][64][32];             // [buf][col][k] fragment-ready
  __shared__ half_t aT[2][4][16][32];          // [buf][wave][m][k]
  __shared__ int    colInfo[64];               // packed (n<<20)|(oy<<10)|ox

  const int tid   = threadIdx.x;
  const int wave  = tid >> 5;
  const int lane  = tid & 31;
  const int hhalf = lane >> 4;
  const int lo    = lane & 15;

  const int colBase = blockIdx.x * 64;
  const int mBase   = blockIdx.y * 256 + wave * 16;
  const bool mValid = (mBase < Cout);
  const int HW      = Hout * Wout;

  if (tid < 64) {
    int col = colBase + tid;
    int n   = col / HW;
    int r   = col - n * HW;
    int oy  = r / Wout;
    int ox  = r - oy * Wout;
    colInfo[tid] = (n << 20) | (oy << 10) | ox;
  }
  __syncthreads();

  auto stage = [&](int s, int pb) {
    int tap = s / CCH;                         // constexpr divisor -> shift
    int ch  = s - tap * CCH;
    int kh  = tap / 3;
    int kw  = tap - kh * 3;
    // ---- B: thread -> column c = tid>>1, K half kh16 = (tid&1)*16
    {
      int c    = tid >> 1;
      int kh16 = (tid & 1) * 16;
      int info = colInfo[c];
      int ox = info & 1023, oy = (info >> 10) & 1023, n = info >> 20;
      int iy, ix; bool valid;
      if (MODE == CONV_FWD) {
        iy = oy * 2 - 1 + kh;
        ix = ox * 2 - 1 + kw;
        valid = ((unsigned)iy < (unsigned)Hin) && ((unsigned)ix < (unsigned)Win);
      } else {
        int ty = oy + 1 - kh, tx = ox + 1 - kw;
        valid = (ty >= 0) && (tx >= 0) && !(ty & 1) && !(tx & 1);
        iy = ty >> 1; ix = tx >> 1;
        valid = valid && (iy < Hin) && (ix < Win);
      }
      int ciBase = ch * 32 + kh16;
      const half_t* src = x + ((size_t)(n * Hin + iy) * Win + ix) * CIN + ciBase;
      bool full = FULLC || (ciBase + 15 < CIN);
#if HAVE_ASYNC_LDS
      if (valid && full) {                     // async b128 x2, no VGPR bounce
        __builtin_amdgcn_global_load_async_to_lds_b128(
            AS1P(src),     AS3P(&bT[pb][c][kh16]),     0, 0);
        __builtin_amdgcn_global_load_async_to_lds_b128(
            AS1P(src + 8), AS3P(&bT[pb][c][kh16 + 8]), 0, 0);
      } else
#endif
      {
        v8h b0 = {0,0,0,0,0,0,0,0}, b1 = {0,0,0,0,0,0,0,0};
        if (valid && full) {
          b0 = *(const v8h*)(src);
          b1 = *(const v8h*)(src + 8);
        } else if (valid) {                    // ragged channel tail (CIN=3)
          #pragma unroll
          for (int i = 0; i < 8; ++i) {
            if (ciBase + i     < CIN) b0[i] = src[i];
            if (ciBase + 8 + i < CIN) b1[i] = src[8 + i];
          }
        }
        *(v8h*)&bT[pb][c][kh16]     = b0;
        *(v8h*)&bT[pb][c][kh16 + 8] = b1;
      }
    }
    // ---- A: lane -> m = lane>>1, K half = (lane&1)*16
    if (mValid) {
      int m    = lane >> 1;
      int kh16 = (lane & 1) * 16;
      int row  = mBase + m;
      size_t wo = (size_t)row * KP + (size_t)s * 32 + kh16;
      if (row < Cout) {
#if HAVE_ASYNC_LDS
        __builtin_amdgcn_global_load_async_to_lds_b128(
            AS1P(wpk + wo),     AS3P(&aT[pb][wave][m][kh16]),     0, 0);
        __builtin_amdgcn_global_load_async_to_lds_b128(
            AS1P(wpk + wo + 8), AS3P(&aT[pb][wave][m][kh16 + 8]), 0, 0);
#else
        v8h a0 = *(const v8h*)(wpk + wo);
        v8h a1 = *(const v8h*)(wpk + wo + 8);
        *(v8h*)&aT[pb][wave][m][kh16]     = a0;
        *(v8h*)&aT[pb][wave][m][kh16 + 8] = a1;
#endif
        __builtin_prefetch((const void*)(wpk + wo + 64), 0, 1); // s+2 stream
      } else {
        v8h z = {0,0,0,0,0,0,0,0};
        *(v8h*)&aT[pb][wave][m][kh16]     = z;
        *(v8h*)&aT[pb][wave][m][kh16 + 8] = z;
      }
    }
  };

  v8f acc[4];
  #pragma unroll
  for (int nt = 0; nt < 4; ++nt) acc[nt] = (v8f){0.f,0.f,0.f,0.f,0.f,0.f,0.f,0.f};

  stage(0, 0);
#if HAVE_ASYNC_LDS
  ASYNC_WAIT();
#endif
  __syncthreads();

  for (int s = 0; s < NSTEPS; ++s) {
    int cur = s & 1;
    if (s + 1 < NSTEPS) stage(s + 1, cur ^ 1);
    if (mValid) {
      // A fragment: k = half*16 + e (matches B)
      v8h a0 = *(const v8h*)&aT[cur][wave][lo][hhalf * 16];
      v8h a1 = *(const v8h*)&aT[cur][wave][lo][hhalf * 16 + 8];
      v16h af = __builtin_shufflevector(a0, a1, 0,1,2,3,4,5,6,7,8,9,10,11,12,13,14,15);
      #pragma unroll
      for (int nt = 0; nt < 4; ++nt) {
        int c  = nt * 16 + lo;
        // B fragment (contiguous k order, same as A): col = lane&15, k = half*16 + e
        v8h b0 = *(const v8h*)&bT[cur][c][hhalf * 16];
        v8h b1 = *(const v8h*)&bT[cur][c][hhalf * 16 + 8];
        v16h bf = __builtin_shufflevector(b0, b1, 0,1,2,3,4,5,6,7,8,9,10,11,12,13,14,15);
        acc[nt] = __builtin_amdgcn_wmma_f32_16x16x32_f16(
                      false, af, false, bf, (short)0, acc[nt], false, false);
      }
    }
#if HAVE_ASYNC_LDS
    ASYNC_WAIT();              // producer drains its async loads pre-barrier
#endif
    __syncthreads();
  }

  if (!mValid) return;

  // D layout: M = 8*half + vgpr, N = lane&15
  const float* bp = bias + mBase + hhalf * 8;
  #pragma unroll
  for (int nt = 0; nt < 4; ++nt) {
    int c    = nt * 16 + lo;
    int info = colInfo[c];
    int ox = info & 1023, oy = (info >> 10) & 1023, n = info >> 20;
    size_t pix = (size_t)(n * Hout + oy) * Wout + ox;
    if (EPI == EPI_RELU_F16) {
      v8h ov;
      #pragma unroll
      for (int rr = 0; rr < 8; ++rr) {
        float v = acc[nt][rr] + bp[rr];
        ov[rr] = (half_t)(v > 0.f ? v : 0.f);
      }
      *(v8h*)&out16[pix * Cout + mBase + hhalf * 8] = ov;
    } else if (EPI == EPI_LIN_F32) {
      v8f ov;
      #pragma unroll
      for (int rr = 0; rr < 8; ++rr) ov[rr] = acc[nt][rr] + bp[rr];
      *(v8f*)&out32[pix * Cout + mBase + hhalf * 8] = ov;
    } else { // double_relu -> NCHW f32 final output
      #pragma unroll
      for (int rr = 0; rr < 8; ++rr) {
        int co = mBase + hhalf * 8 + rr;
        if (co < Cout) {
          float v = acc[nt][rr] + bias[co];
          float y = (v < 0.f) ? 0.01f * v
                  : (v > 1.f) ? 1.f + 0.01f * (v - 1.f)
                              : v;
          out32[((size_t)(n * Cout + co) * Hout + oy) * Wout + ox] = y;
        }
      }
    }
  }
}

// ---------------------------------------------------------------------------
// Weight packing: f32 -> f16 [Cout][Kpad], tap-major K: k = (tap*CCH+ch)*32+kk,
// ci = ch*32+kk (zero-padded past Cin). trans=1 flips + transposes the
// ConvTranspose2d weight [Cin][Cout][3][3].
// ---------------------------------------------------------------------------
__global__ void pack_w_k(const float* __restrict__ w, half_t* __restrict__ wpk,
                         int Cout, int Cin, int CCH, int Kpad, int trans)
{
  int idx = blockIdx.x * blockDim.x + threadIdx.x;
  if (idx >= Cout * Kpad) return;
  int co   = idx / Kpad;
  int kidx = idx - co * Kpad;
  int s    = kidx >> 5;
  int kk   = kidx & 31;
  int tap  = s / CCH;
  int ch   = s - tap * CCH;
  int ci   = ch * 32 + kk;
  float v = 0.f;
  if (ci < Cin) {
    if (trans) {
      int kh = tap / 3, kw = tap - kh * 3;
      v = w[((size_t)(ci * Cout + co) * 3 + (2 - kh)) * 3 + (2 - kw)];
    } else {
      v = w[((size_t)co * Cin + ci) * 9 + tap];
    }
  }
  wpk[idx] = (half_t)v;
}

// NCHW f32 -> NHWC f16
__global__ void nchw_to_nhwc_k(const float* __restrict__ in, half_t* __restrict__ out,
                               int N, int C, int H, int W)
{
  int idx = blockIdx.x * blockDim.x + threadIdx.x;
  int total = N * C * H * W;
  if (idx >= total) return;
  int c = idx % C; int t = idx / C;
  int xx = t % W;  t /= W;
  int yy = t % H;  int n = t / H;
  out[idx] = (half_t)in[((size_t)(n * C + c) * H + yy) * W + xx];
}

// one-time table of center L2 norms: nrm[c][k] = ||centers[c][:,k]||
__global__ void center_norms_k(const float* __restrict__ centers,
                               float* __restrict__ nrm, int C)
{
  int idx = blockIdx.x * blockDim.x + threadIdx.x;
  if (idx >= C * 256) return;
  int c = idx >> 8, k = idx & 255;
  float s = 0.f;
  #pragma unroll
  for (int j = 0; j < 4; ++j) {
    float w = centers[(size_t)(c * 4 + j) * 256 + k];
    s += w * w;
  }
  nrm[idx] = sqrtf(s);
}

// ---------------------------------------------------------------------------
// Patch quantizer (NHWC): per 2x2 patch, cosine sim vs 256 centers, softmax
// mix. Lanes of a wave share the channel c, so the center stream is
// wave-uniform (broadcast-friendly); norms come from the precomputed table.
// ---------------------------------------------------------------------------
__global__ void quantize_k(const float* __restrict__ h,        // NHWC f32
                           const float* __restrict__ centers,  // [C][4][256]
                           const float* __restrict__ nrm,      // [C][256]
                           const float* __restrict__ sigma,
                           half_t* __restrict__ q,             // NHWC f16
                           int N, int C, int H, int W)
{
  int H2 = H >> 1, W2 = W >> 1;
  int total = N * C * H2 * W2;
  int idx = blockIdx.x * blockDim.x + threadIdx.x;
  if (idx >= total) return;
  int px = idx % W2; int t = idx / W2;
  int py = t % H2;   t /= H2;
  int c  = t % C;    int n = t / C;

  const float* base = h + ((size_t)(n * H + 2 * py) * W + 2 * px) * C + c;
  float p0 = base[0], p1 = base[C], p2 = base[(size_t)W * C], p3 = base[(size_t)(W + 1) * C];
  float nx = sqrtf(p0 * p0 + p1 * p1 + p2 * p2 + p3 * p3);
  float sg = sigma[0];

  const float* c0 = centers + (size_t)(c * 4 + 0) * 256;
  const float* c1 = centers + (size_t)(c * 4 + 1) * 256;
  const float* c2 = centers + (size_t)(c * 4 + 2) * 256;
  const float* c3 = centers + (size_t)(c * 4 + 3) * 256;
  const float* cn = nrm + (size_t)c * 256;

  float den = 0.f, a0 = 0.f, a1 = 0.f, a2 = 0.f, a3 = 0.f;
  for (int k = 0; k < 256; ++k) {
    float w0 = c0[k], w1 = c1[k], w2 = c2[k], w3 = c3[k];
    float dot = p0 * w0 + p1 * w1 + p2 * w2 + p3 * w3;
    float l   = dot / fmaxf(nx * cn[k], 1e-8f);
    float e   = __expf(l * sg);
    den += e;
    a0 += e * w0; a1 += e * w1; a2 += e * w2; a3 += e * w3;
  }
  float inv = 1.f / den;
  half_t* qb = q + ((size_t)(n * H + 2 * py) * W + 2 * px) * C + c;
  qb[0]                   = (half_t)(a0 * inv);
  qb[C]                   = (half_t)(a1 * inv);
  qb[(size_t)W * C]       = (half_t)(a2 * inv);
  qb[(size_t)(W + 1) * C] = (half_t)(a3 * inv);
}

// ---------------------------------------------------------------------------
extern "C" void kernel_launch(void* const* d_in, const int* in_sizes, int n_in,
                              void* d_out, int out_size, void* d_ws, size_t ws_size,
                              hipStream_t stream)
{
  const float* img    = (const float*)d_in[0];
  const float* sigma  = (const float*)d_in[1];
  const float* enc_w1 = (const float*)d_in[2];
  const float* enc_b1 = (const float*)d_in[3];
  const float* enc_w2 = (const float*)d_in[4];
  const float* enc_b2 = (const float*)d_in[5];
  const float* enc_w3 = (const float*)d_in[6];
  const float* enc_b3 = (const float*)d_in[7];
  const float* centers= (const float*)d_in[8];
  const float* dec_w1 = (const float*)d_in[9];
  const float* dec_b1 = (const float*)d_in[10];
  const float* dec_w2 = (const float*)d_in[11];
  const float* dec_b2 = (const float*)d_in[12];
  const float* dec_w3 = (const float*)d_in[13];
  const float* dec_b3 = (const float*)d_in[14];
  float* out = (float*)d_out;

  const int NB = 16;

  char* ws = (char*)d_ws;
  size_t off = 0;
  auto alloc = [&](size_t bytes) -> char* {
    char* p = ws + off;
    off = (off + bytes + 255) & ~(size_t)255;
    return p;
  };

  half_t* img16 = (half_t*)alloc((size_t)NB * 256 * 256 * 3 * 2);
  half_t* h1    = (half_t*)alloc((size_t)NB * 128 * 128 * 64 * 2);  // reused as g2
  half_t* h2    = (half_t*)alloc((size_t)NB * 64 * 64 * 128 * 2);   // reused as g1
  float*  h3    = (float*) alloc((size_t)NB * 32 * 32 * 64 * 4);
  half_t* q16   = (half_t*)alloc((size_t)NB * 32 * 32 * 64 * 2);
  float*  nrm   = (float*) alloc((size_t)64 * 256 * 4);
  half_t* wpk1  = (half_t*)alloc((size_t)64  * 288  * 2);
  half_t* wpk2  = (half_t*)alloc((size_t)128 * 576  * 2);
  half_t* wpk3  = (half_t*)alloc((size_t)64  * 1152 * 2);
  half_t* wpkd1 = (half_t*)alloc((size_t)128 * 576  * 2);
  half_t* wpkd2 = (half_t*)alloc((size_t)64  * 1152 * 2);
  half_t* wpkd3 = (half_t*)alloc((size_t)3   * 576  * 2);
  half_t* g1 = h2;   // h2 dead after conv3
  half_t* g2 = h1;   // h1 dead after conv2

  // ---- weight packing + input layout conversion + norm table
  {
    int n;
    n = 64 * 288;   pack_w_k<<<(n + 255) / 256, 256, 0, stream>>>(enc_w1, wpk1, 64, 3, 1, 288, 0);
    n = 128 * 576;  pack_w_k<<<(n + 255) / 256, 256, 0, stream>>>(enc_w2, wpk2, 128, 64, 2, 576, 0);
    n = 64 * 1152;  pack_w_k<<<(n + 255) / 256, 256, 0, stream>>>(enc_w3, wpk3, 64, 128, 4, 1152, 0);
    n = 128 * 576;  pack_w_k<<<(n + 255) / 256, 256, 0, stream>>>(dec_w1, wpkd1, 128, 64, 2, 576, 1);
    n = 64 * 1152;  pack_w_k<<<(n + 255) / 256, 256, 0, stream>>>(dec_w2, wpkd2, 64, 128, 4, 1152, 1);
    n = 3 * 576;    pack_w_k<<<(n + 255) / 256, 256, 0, stream>>>(dec_w3, wpkd3, 3, 64, 2, 576, 1);
    n = NB * 3 * 256 * 256;
    nchw_to_nhwc_k<<<(n + 255) / 256, 256, 0, stream>>>(img, img16, NB, 3, 256, 256);
    n = 64 * 256;
    center_norms_k<<<(n + 255) / 256, 256, 0, stream>>>(centers, nrm, 64);
  }

  const dim3 blk(128);

  // ---- encoder (NHWC f16 activations)
  conv_wmma_k<CONV_FWD, EPI_RELU_F16, 3><<<dim3(NB * 128 * 128 / 64, 1), blk, 0, stream>>>(
      img16, wpk1, enc_b1, h1, nullptr, 256, 256, 64, 128, 128);
  conv_wmma_k<CONV_FWD, EPI_RELU_F16, 64><<<dim3(NB * 64 * 64 / 64, 2), blk, 0, stream>>>(
      h1, wpk2, enc_b2, h2, nullptr, 128, 128, 128, 64, 64);
  conv_wmma_k<CONV_FWD, EPI_LIN_F32, 128><<<dim3(NB * 32 * 32 / 64, 1), blk, 0, stream>>>(
      h2, wpk3, enc_b3, nullptr, h3, 64, 64, 64, 32, 32);

  // ---- quantizer
  {
    int n = NB * 64 * 16 * 16;
    quantize_k<<<(n + 255) / 256, 256, 0, stream>>>(h3, centers, nrm, sigma, q16, NB, 64, 32, 32);
  }

  // ---- decoder
  conv_wmma_k<CONV_TRANS, EPI_RELU_F16, 64><<<dim3(NB * 64 * 64 / 64, 2), blk, 0, stream>>>(
      q16, wpkd1, dec_b1, g1, nullptr, 32, 32, 128, 64, 64);
  conv_wmma_k<CONV_TRANS, EPI_RELU_F16, 128><<<dim3(NB * 128 * 128 / 64, 1), blk, 0, stream>>>(
      g1, wpkd2, dec_b2, g2, nullptr, 64, 64, 64, 128, 128);
  conv_wmma_k<CONV_TRANS, EPI_DRELU_F32, 64><<<dim3(NB * 256 * 256 / 64, 1), blk, 0, stream>>>(
      g2, wpkd3, dec_b3, nullptr, out, 128, 128, 3, 256, 256);

  (void)in_sizes; (void)n_in; (void)out_size; (void)ws_size;
}
